// RaBiT_Fusion_11166914969944
// MI455X (gfx1250) — matrix operand order repaired and checked by reading, not compiled, not run on hardware
//
#include <hip/hip_runtime.h>
#include <math.h>

typedef _Float16 h16;
typedef __attribute__((ext_vector_type(8)))  _Float16 v8h;
typedef __attribute__((ext_vector_type(16))) _Float16 v16h;
typedef __attribute__((ext_vector_type(8)))  float    v8f;

#define SCALE 0.17677669529663687f  /* 32^-0.5 */

__device__ __forceinline__ v16h cat8(v8h lo, v8h hi) {
  return __builtin_shufflevector(lo, hi, 0,1,2,3,4,5,6,7,8,9,10,11,12,13,14,15);
}

// A-fragment, 16x32 f16, row-major source with leading dim lda (halfs).
// ISA layout: lanes 0-15 -> rows M=0..15, K {0..7,16..23}; lanes 16-31 -> K {8..15,24..31}.
__device__ __forceinline__ v16h load_afrag(const h16* base, int lda, int row0, int k0, int lane) {
  int r  = row0 + (lane & 15);
  int kb = k0 + ((lane >> 4) << 3);
  const h16* p = base + (size_t)r * lda + kb;
  v8h lo = *(const v8h*)(p);
  v8h hi = *(const v8h*)(p + 16);
  return cat8(lo, hi);
}

// B-fragment from pre-packed weights: per (kt,nt) tile, 32 lanes x 16 halfs contiguous.
__device__ __forceinline__ v16h load_bfrag(const h16* Bp, int ntTot, int kt, int nt, int lane) {
  return *(const v16h*)(Bp + (((size_t)kt * ntTot + nt) * 32 + lane) * 16);
}

__device__ __forceinline__ v8f wmma16(v16h a, v16h b, v8f c) {
  return __builtin_amdgcn_wmma_f32_16x16x32_f16(false, a, false, b, (short)0, c, false, false);
}

__device__ __forceinline__ float wred32(float v) {
#pragma unroll
  for (int o = 16; o > 0; o >>= 1) v += __shfl_xor(v, o, 32);
  return v;
}

// ---------------------------------------------------------------------------
// Pack f32 weight (K x N row-major) into WMMA B-fragment order, f16.
__global__ __launch_bounds__(256) void k_pack(const float* __restrict__ W,
                                              h16* __restrict__ Bp, int K, int N) {
  int idx = blockIdx.x * 256 + threadIdx.x;
  if (idx >= K * N) return;
  int k = idx / N, n = idx - k * N;
  int kt = k >> 5, kr = k & 31, nt = n >> 4, nr = n & 15;
  int lane = nr + ((kr & 8) << 1);
  int slot = (kr & 7) | ((kr >> 1) & 8);
  int NT = N >> 4;
  Bp[(((size_t)kt * NT + nt) * 32 + lane) * 16 + slot] = (h16)W[idx];
}

// ---------------------------------------------------------------------------
// gamma = sigmoid(relu([meanR,meanT] @ g1 + b1) @ g2 + b2), per batch.
__global__ __launch_bounds__(256) void k_gamma(const float* rR, const float* rT,
    const float* g1w, const float* g1b, const float* g2w, const float* g2b, float* gOut) {
  int b = blockIdx.x, tid = threadIdx.x;
  __shared__ float red[16];
  const float* pR = rR + (size_t)b * 16384;
  const float* pT = rT + (size_t)b * 16384;
  float sR = 0.f, sT = 0.f;
  for (int i = tid; i < 16384; i += 256) { sR += pR[i]; sT += pT[i]; }
  sR = wred32(sR); sT = wred32(sT);
  int w = tid >> 5, lane = tid & 31;
  if (lane == 0) { red[w * 2] = sR; red[w * 2 + 1] = sT; }
  __syncthreads();
  if (tid == 0) {
    float aR = 0.f, aT = 0.f;
    for (int i = 0; i < 8; ++i) { aR += red[i * 2]; aT += red[i * 2 + 1]; }
    aR *= (1.f / 16384.f); aT *= (1.f / 16384.f);
    float s = g2b[0];
    for (int j = 0; j < 8; ++j) {
      float hj = aR * g1w[j] + aT * g1w[8 + j] + g1b[j];
      s += fmaxf(hj, 0.f) * g2w[j];
    }
    gOut[b] = 1.f / (1.f + expf(-s));
  }
}

// ---------------------------------------------------------------------------
// Gather windows from NCHW, write window-major f16, accumulate concat-LN stats.
__global__ __launch_bounds__(256) void k_winprep(const float* __restrict__ FR,
    const float* __restrict__ FT, const float* __restrict__ rR, const float* __restrict__ rT,
    h16* __restrict__ FR16, h16* __restrict__ FT16, float* __restrict__ rowStat,
    float* __restrict__ rRw, float* __restrict__ rTw) {
  int wid = blockIdx.x, tid = threadIdx.x;
  int b = wid >> 8, wh = (wid >> 4) & 15, ww = wid & 15;
  int h0 = wh << 3, w0 = ww << 3;
  __shared__ h16 sBuf[64 * 264];
  __shared__ float sSum[64], sSq[64];
  if (tid < 64) { sSum[tid] = 0.f; sSq[tid] = 0.f; }
  int wv = tid >> 5, lane = tid & 31;
  int seg = tid >> 3, wq = tid & 7;
  for (int pass = 0; pass < 2; ++pass) {
    const float* src = pass ? FT : FR;
    h16* dst = pass ? FT16 : FR16;
    __syncthreads();
    for (int i = 0; i < 64; ++i) {     // gather (8 contiguous w per 8 threads)
      int lin = i * 32 + seg;
      int c = lin >> 3, hr = lin & 7, n = (hr << 3) + wq;
      float v = src[(((size_t)b * 256 + c) * 128 + (h0 + hr)) * 128 + (w0 + wq)];
      sBuf[n * 264 + c] = (h16)v;
    }
    __syncthreads();
    for (int j = 0; j < 8; ++j) {      // per-row partial sums for concat-LN
      int n = wv * 8 + j;
      float s = 0.f, q = 0.f;
#pragma unroll
      for (int k = 0; k < 8; ++k) {
        float v = (float)sBuf[n * 264 + lane + 32 * k];
        s += v; q += v * v;
      }
      s = wred32(s); q = wred32(q);
      if (lane == 0) { sSum[n] += s; sSq[n] += q; }
    }
    for (int i = 0; i < 8; ++i) {      // coalesced f16 write-out
      int flat = tid + 256 * i;
      int n = flat >> 5, c8 = (flat & 31) << 3;
      *(v8h*)(dst + ((size_t)wid * 64 + n) * 256 + c8) = *(const v8h*)(sBuf + n * 264 + c8);
    }
  }
  __syncthreads();
  if (tid < 64) {
    float m = sSum[tid] * (1.f / 512.f);
    float var = sSq[tid] * (1.f / 512.f) - m * m;
    rowStat[((size_t)wid * 64 + tid) * 2]     = m;
    rowStat[((size_t)wid * 64 + tid) * 2 + 1] = rsqrtf(var + 1e-5f);
    int hr = tid >> 3; int wq2 = tid & 7;
    size_t ra = ((size_t)b * 128 + (h0 + hr)) * 128 + (w0 + wq2);
    rRw[wid * 64 + tid] = rR[ra];
    rTw[wid * 64 + tid] = rT[ra];
  }
}

// ---------------------------------------------------------------------------
// logits = F_hat@psi+b, softmax over n; Y = A^T F_hat (4x512) written f16.
__global__ __launch_bounds__(256) void k_psiA(const h16* __restrict__ FR16,
    const h16* __restrict__ FT16, const float* __restrict__ rowStat,
    const float* __restrict__ psi_w, const float* __restrict__ psi_b,
    const float* __restrict__ np_g, const float* __restrict__ np_b, h16* __restrict__ Y16) {
  int wid = blockIdx.x, tid = threadIdx.x;
  __shared__ float sPsi[2048], sNg[512], sNb[512];
  __shared__ float sMean[64], sInv[64], sA[256], sMx[4], sSm[4];
  for (int i = tid; i < 2048; i += 256) sPsi[i] = psi_w[i];
  for (int i = tid; i < 512; i += 256) { sNg[i] = np_g[i]; sNb[i] = np_b[i]; }
  if (tid < 64) {
    sMean[tid] = rowStat[((size_t)wid * 64 + tid) * 2];
    sInv[tid]  = rowStat[((size_t)wid * 64 + tid) * 2 + 1];
  }
  __syncthreads();
  {
    int n = tid >> 2, m = tid & 3;
    const h16* rowR = FR16 + ((size_t)wid * 64 + n) * 256;
    const h16* rowT = FT16 + ((size_t)wid * 64 + n) * 256;
    float mean = sMean[n], inv = sInv[n];
    float s = psi_b[m];
    for (int d8 = 0; d8 < 32; ++d8) {
      v8h a = *(const v8h*)(rowR + d8 * 8);
      v8h t = *(const v8h*)(rowT + d8 * 8);
#pragma unroll
      for (int j = 0; j < 8; ++j) {
        int d = d8 * 8 + j;
        float fr = ((float)a[j] - mean) * inv * sNg[d] + sNb[d];
        float ft = ((float)t[j] - mean) * inv * sNg[256 + d] + sNb[256 + d];
        s += fr * sPsi[d * 4 + m] + ft * sPsi[(256 + d) * 4 + m];
      }
    }
    sA[n * 4 + m] = s;
  }
  __syncthreads();
  if (tid < 4) {
    float mx = -1e30f;
    for (int n = 0; n < 64; ++n) mx = fmaxf(mx, sA[n * 4 + tid]);
    float sm = 0.f;
    for (int n = 0; n < 64; ++n) sm += expf(sA[n * 4 + tid] - mx);
    sMx[tid] = mx; sSm[tid] = 1.f / sm;
  }
  __syncthreads();
  { int n = tid >> 2, m = tid & 3;
    sA[n * 4 + m] = expf(sA[n * 4 + m] - sMx[m]) * sSm[m]; }
  __syncthreads();
  for (int i = 0; i < 8; ++i) {
    int idx = tid + 256 * i;
    int m = idx >> 9, d = idx & 511;
    float g = sNg[d], bb = sNb[d];
    const h16* base = (d < 256) ? (FR16 + (size_t)wid * 64 * 256 + d)
                                : (FT16 + (size_t)wid * 64 * 256 + (d - 256));
    float y = 0.f;
    for (int n = 0; n < 64; ++n) {
      float fh = ((float)base[n * 256] - sMean[n]) * sInv[n] * g + bb;
      y += sA[n * 4 + m] * fh;
    }
    Y16[((size_t)wid * 4 + m) * 512 + d] = (h16)y;
  }
}

// ---------------------------------------------------------------------------
// Batched GEMM: A [M][K] f16 row-major, B pre-packed, out f16 (+bias, opt GELU).
// Block tile 64(M) x 128(N); 8 waves, 4 wmma tiles/wave, A staged in LDS.
// kt loop kept rolled so only one iteration's fragments are in flight (no spills).
template<int ACT>
__global__ __launch_bounds__(256) void k_gemm(const h16* __restrict__ A,
    const h16* __restrict__ Bp, const float* __restrict__ bias, h16* __restrict__ Out,
    int M, int K, int N) {
  __shared__ h16 sA[64 * 40];
  int tid = threadIdx.x, w = tid >> 5, lane = tid & 31;
  int m0 = blockIdx.y * 64;
  int nt0 = blockIdx.x * 8;
  int NTtot = N >> 4;
  int mt = w & 3;
  int ntW = (w >> 2) * 4;
  int KT = K >> 5;
  v8f acc[4] = {};
  int srow = tid >> 2, scol = (tid & 3) << 3;
#pragma unroll 1
  for (int kt = 0; kt < KT; ++kt) {
    __syncthreads();
    *(v8h*)(sA + srow * 40 + scol) =
        *(const v8h*)(A + (size_t)(m0 + srow) * K + (kt << 5) + scol);
    __syncthreads();
    v16h af = load_afrag(sA, 40, mt * 16, 0, lane);
#pragma unroll
    for (int j = 0; j < 4; ++j) {
      v16h bf = load_bfrag(Bp, NTtot, kt, nt0 + ntW + j, lane);
      acc[j] = wmma16(af, bf, acc[j]);
    }
    if (kt + 1 < KT)
      __builtin_prefetch(Bp + (((size_t)(kt + 1) * NTtot + nt0 + ntW) * 32 + lane) * 16, 0, 3);
  }
  int rowBase = m0 + mt * 16 + ((lane >> 4) << 3);
#pragma unroll
  for (int j = 0; j < 4; ++j) {
    int col = (nt0 + ntW + j) * 16 + (lane & 15);
    float bv = bias ? bias[col] : 0.f;
#pragma unroll
    for (int r = 0; r < 8; ++r) {
      float v = acc[j][r] + bv;
      if (ACT == 1) v = 0.5f * v * (1.f + erff(v * 0.70710678118654752f));
      Out[(size_t)(rowBase + r) * N + col] = (h16)v;
    }
  }
}

// ---------------------------------------------------------------------------
// In-window projection GEMM: (64x256)@(256x256)+bias -> LDS [64][264] f16.
// A rows loaded straight from L2-resident window-major f16 buffers.
// Processed as two groups of 4 N-tiles (32 acc VGPRs live at a time) with the
// K loop rolled, so nothing spills to scratch.
__device__ void win_gemm(const h16* __restrict__ Aglob, const h16* __restrict__ Bp,
                         const float* __restrict__ bias, h16* sOut, int wid) {
  int tid = threadIdx.x, w = tid >> 5, lane = tid & 31;
  int mt = w >> 1;
  int ntBase = (w & 1) * 8;
  const h16* A0 = Aglob + (size_t)wid * 64 * 256;
  int rowBase = mt * 16 + ((lane >> 4) << 3);
#pragma unroll 1
  for (int g = 0; g < 2; ++g) {
    int nt0 = ntBase + g * 4;
    v8f acc[4] = {};
#pragma unroll 1
    for (int kt = 0; kt < 8; ++kt) {
      v16h af = load_afrag(A0, 256, mt * 16, kt * 32, lane);
#pragma unroll
      for (int j = 0; j < 4; ++j) {
        v16h bf = load_bfrag(Bp, 16, kt, nt0 + j, lane);
        acc[j] = wmma16(af, bf, acc[j]);
      }
    }
#pragma unroll
    for (int j = 0; j < 4; ++j) {
      int col = (nt0 + j) * 16 + (lane & 15);
      float bv = bias[col];
#pragma unroll
      for (int r = 0; r < 8; ++r)
        sOut[(rowBase + r) * 264 + col] = (h16)(acc[j][r] + bv);
    }
  }
}

// ---------------------------------------------------------------------------
// Z = LN((Zphi[:, :256] + Zphi[:, 256:]) * 0.5)
__global__ __launch_bounds__(128) void k_Zln(const h16* __restrict__ Zphi,
    const float* nz_g, const float* nz_b, h16* __restrict__ Z16) {
  int wid = blockIdx.x, tid = threadIdx.x;
  int m = tid >> 5, lane = tid & 31;
  const h16* row = Zphi + ((size_t)wid * 4 + m) * 512;
  float v[8]; float s = 0.f, q = 0.f;
#pragma unroll
  for (int k = 0; k < 8; ++k) {
    int c = lane + 32 * k;
    float z = ((float)row[c] + (float)row[256 + c]) * 0.5f;
    v[k] = z; s += z; q += z * z;
  }
  s = wred32(s); q = wred32(q);
  float mean = s * (1.f / 256.f);
  float inv = rsqrtf(q * (1.f / 256.f) - mean * mean + 1e-5f);
  h16* orow = Z16 + ((size_t)wid * 4 + m) * 256;
#pragma unroll
  for (int k = 0; k < 8; ++k) {
    int c = lane + 32 * k;
    orow[c] = (h16)((v[k] - mean) * inv * nz_g[c] + nz_b[c]);
  }
}

// ---------------------------------------------------------------------------
// Cross-attn: K/V projections in-kernel (WMMA), softmax over n, Zp = LN(Z+ZhR+ZhT).
__global__ __launch_bounds__(256) void k_xattn(const h16* __restrict__ FR16,
    const h16* __restrict__ FT16, const h16* __restrict__ kP, const h16* __restrict__ vP,
    const float* k_b, const float* v_b, const h16* __restrict__ QZ16,
    const h16* __restrict__ Z16, const float* __restrict__ rRw, const float* __restrict__ rTw,
    const float* nzf_g, const float* nzf_b, h16* __restrict__ Zp16) {
  int wid = blockIdx.x, tid = threadIdx.x;
  __shared__ h16 sBuf[64 * 264];
  __shared__ float sQZ[1024], sZres[1024], sZacc[1024];
  __shared__ float sLog[2048];
  __shared__ float sMS[64];
  __shared__ float sRv[128];
  for (int i = 0; i < 4; ++i) {
    int idx = tid + 256 * i;
    sQZ[idx]   = (float)QZ16[(size_t)wid * 4 * 256 + idx];
    sZres[idx] = (float)Z16[(size_t)wid * 4 * 256 + idx];
    sZacc[idx] = 0.f;
  }
  if (tid < 64) { sRv[tid] = rRw[wid * 64 + tid]; sRv[64 + tid] = rTw[wid * 64 + tid]; }
  __syncthreads();
  for (int br = 0; br < 2; ++br) {
    const h16* F = br ? FT16 : FR16;
    const float* rv = br ? (sRv + 64) : sRv;
    win_gemm(F, kP, k_b, sBuf, wid);           // K projection
    __syncthreads();
    for (int i = 0; i < 8; ++i) {              // logits[h][m][n]
      int idx = tid + 256 * i;
      int hh = idx >> 8, rem = idx & 255, m = rem >> 6, n = rem & 63;
      float s = 0.f;
#pragma unroll
      for (int d = 0; d < 32; ++d)
        s += sQZ[m * 256 + hh * 32 + d] * (float)sBuf[n * 264 + hh * 32 + d];
      sLog[idx] = s * SCALE * rv[n];
    }
    __syncthreads();
    if (tid < 32) {                            // softmax over n
      int base = tid * 64;
      float mx = -1e30f;
      for (int n = 0; n < 64; ++n) mx = fmaxf(mx, sLog[base + n]);
      float sm = 0.f;
      for (int n = 0; n < 64; ++n) sm += expf(sLog[base + n] - mx);
      sMS[tid * 2] = mx; sMS[tid * 2 + 1] = 1.f / sm;
    }
    __syncthreads();
    for (int i = 0; i < 8; ++i) {
      int idx = tid + 256 * i;
      int hm = idx >> 6;
      sLog[idx] = expf(sLog[idx] - sMS[hm * 2]) * sMS[hm * 2 + 1];
    }
    __syncthreads();
    win_gemm(F, vP, v_b, sBuf, wid);           // V projection (reuse buffer)
    __syncthreads();
    for (int i = 0; i < 4; ++i) {              // Zh accumulate
      int idx = tid + 256 * i;
      int m = idx >> 8, c = idx & 255, hh = c >> 5;
      float a = 0.f;
      for (int n = 0; n < 64; ++n)
        a += sLog[(hh * 4 + m) * 64 + n] * (float)sBuf[n * 264 + c];
      sZacc[idx] += a;
    }
    __syncthreads();
  }
  int w = tid >> 5, lane = tid & 31;
  if (w < 4) {                                 // Zp = LN(Z + Zh)
    float v[8]; float s = 0.f, q = 0.f;
#pragma unroll
    for (int k = 0; k < 8; ++k) {
      int c = lane + 32 * k;
      float x = sZres[w * 256 + c] + sZacc[w * 256 + c];
      v[k] = x; s += x; q += x * x;
    }
    s = wred32(s); q = wred32(q);
    float mean = s * (1.f / 256.f);
    float inv = rsqrtf(q * (1.f / 256.f) - mean * mean + 1e-5f);
#pragma unroll
    for (int k = 0; k < 8; ++k) {
      int c = lane + 32 * k;
      Zp16[((size_t)wid * 4 + w) * 256 + c] = (h16)((v[k] - mean) * inv * nzf_g[c] + nzf_b[c]);
    }
  }
}

// ---------------------------------------------------------------------------
// Second attention + residual LN + gamma blend. Q projection in-kernel (WMMA);
// attention output overwrites the Q slice in place. br==0 writes, br==1 adds.
__global__ __launch_bounds__(256) void k_sattn(const h16* __restrict__ F16,
    const h16* __restrict__ qP, const float* q_b, const h16* __restrict__ KZ16,
    const h16* __restrict__ VZ16, const float* __restrict__ rwin,
    const float* ng, const float* nb, const float* __restrict__ gammaB,
    float* __restrict__ out, int br) {
  int wid = blockIdx.x, tid = threadIdx.x;
  int b = wid >> 8, wh = (wid >> 4) & 15, ww = wid & 15;
  int h0 = wh << 3, w0 = ww << 3;
  __shared__ h16 sBuf[64 * 264];
  __shared__ float sKZ[1024], sVZ[1024];
  __shared__ float sRv[64];
  __shared__ float sStat[128];
  for (int i = 0; i < 4; ++i) {
    int idx = tid + 256 * i;
    sKZ[idx] = (float)KZ16[(size_t)wid * 4 * 256 + idx];
    sVZ[idx] = (float)VZ16[(size_t)wid * 4 * 256 + idx];
  }
  if (tid < 64) sRv[tid] = rwin[wid * 64 + tid];
  __syncthreads();
  win_gemm(F16, qP, q_b, sBuf, wid);
  __syncthreads();
  for (int i = 0; i < 2; ++i) {                // per-(n,h) slice, softmax over m=4
    int idx = tid + 256 * i;
    int n = idx >> 3, hh = idx & 7;
    float qv[32];
#pragma unroll
    for (int d = 0; d < 32; ++d) qv[d] = (float)sBuf[n * 264 + hh * 32 + d];
    float lg[4];
#pragma unroll
    for (int m = 0; m < 4; ++m) {
      float s = 0.f;
#pragma unroll
      for (int d = 0; d < 32; ++d) s += qv[d] * sKZ[m * 256 + hh * 32 + d];
      lg[m] = s * SCALE * sRv[n];
    }
    float mx = fmaxf(fmaxf(lg[0], lg[1]), fmaxf(lg[2], lg[3]));
    float e0 = expf(lg[0] - mx), e1 = expf(lg[1] - mx);
    float e2 = expf(lg[2] - mx), e3 = expf(lg[3] - mx);
    float rs = 1.f / (e0 + e1 + e2 + e3);
    e0 *= rs; e1 *= rs; e2 *= rs; e3 *= rs;
#pragma unroll
    for (int d = 0; d < 32; ++d) {
      int c = hh * 32 + d;
      float o = e0 * sVZ[c] + e1 * sVZ[256 + c] + e2 * sVZ[512 + c] + e3 * sVZ[768 + c];
      sBuf[n * 264 + c] = (h16)o;              // in-place: Fh overwrites Q slice
    }
  }
  __syncthreads();
  int w = tid >> 5, lane = tid & 31;
  const h16* Fw = F16 + (size_t)wid * 64 * 256;
  for (int j = 0; j < 8; ++j) {                // LN stats of (F + Fh) rows
    int n = w * 8 + j;
    float s = 0.f, q = 0.f;
#pragma unroll
    for (int k = 0; k < 8; ++k) {
      int c = lane + 32 * k;
      float x = (float)Fw[n * 256 + c] + (float)sBuf[n * 264 + c];
      s += x; q += x * x;
    }
    s = wred32(s); q = wred32(q);
    if (lane == 0) {
      float mean = s * (1.f / 256.f);
      sStat[n * 2] = mean;
      sStat[n * 2 + 1] = rsqrtf(q * (1.f / 256.f) - mean * mean + 1e-5f);
    }
  }
  __syncthreads();
  float gamma = gammaB[b];
  float wgt = br ? (1.f - gamma) : gamma;
  int seg = tid >> 3, wq = tid & 7;
  for (int i = 0; i < 64; ++i) {               // scatter to NCHW (8-wide contiguous w)
    int lin = i * 32 + seg;
    int c = lin >> 3, hr = lin & 7, n = (hr << 3) + wq;
    float x = (float)Fw[n * 256 + c] + (float)sBuf[n * 264 + c];
    float y = (x - sStat[n * 2]) * sStat[n * 2 + 1] * ng[c] + nb[c];
    size_t addr = (((size_t)b * 256 + c) * 128 + (h0 + hr)) * 128 + (w0 + wq);
    if (br == 0) out[addr] = wgt * y;
    else         out[addr] += wgt * y;
  }
}

// ---------------------------------------------------------------------------
extern "C" void kernel_launch(void* const* d_in, const int* in_sizes, int n_in,
                              void* d_out, int out_size, void* d_ws, size_t ws_size,
                              hipStream_t stream) {
  (void)in_sizes; (void)n_in; (void)out_size; (void)ws_size;
  const float* F_R   = (const float*)d_in[0];
  const float* F_T   = (const float*)d_in[1];
  const float* r_R   = (const float*)d_in[2];
  const float* r_T   = (const float*)d_in[3];
  const float* psi_w = (const float*)d_in[4];
  const float* psi_b = (const float*)d_in[5];
  const float* phi_w = (const float*)d_in[6];
  const float* phi_b = (const float*)d_in[7];
  const float* np_g  = (const float*)d_in[8];
  const float* np_b  = (const float*)d_in[9];
  const float* nz_g  = (const float*)d_in[10];
  const float* nz_b  = (const float*)d_in[11];
  /* d_in[12] = rpb_table: unused by the reference */
  const float* q_w   = (const float*)d_in[13];
  const float* q_b   = (const float*)d_in[14];
  const float* k_w   = (const float*)d_in[15];
  const float* k_b   = (const float*)d_in[16];
  const float* v_w   = (const float*)d_in[17];
  const float* v_b   = (const float*)d_in[18];
  const float* nzf_g = (const float*)d_in[19];
  const float* nzf_b = (const float*)d_in[20];
  const float* fc1_w = (const float*)d_in[21];
  const float* fc1_b = (const float*)d_in[22];
  const float* fc2_w = (const float*)d_in[23];
  const float* fc2_b = (const float*)d_in[24];
  const float* nr_g  = (const float*)d_in[25];
  const float* nr_b  = (const float*)d_in[26];
  const float* nt_g  = (const float*)d_in[27];
  const float* nt_b  = (const float*)d_in[28];
  const float* g1_w  = (const float*)d_in[29];
  const float* g1_b  = (const float*)d_in[30];
  const float* g2_w  = (const float*)d_in[31];
  const float* g2_b  = (const float*)d_in[32];
  float* out = (float*)d_out;

  char* ws = (char*)d_ws;
  size_t off = 0;
  auto carve = [&](size_t bytes) -> char* {
    char* p = ws + off;
    off += (bytes + 255) & ~(size_t)255;
    return p;
  };
  h16* qP    = (h16*)carve((size_t)256 * 256 * 2);
  h16* kP    = (h16*)carve((size_t)256 * 256 * 2);
  h16* vP    = (h16*)carve((size_t)256 * 256 * 2);
  h16* phiP  = (h16*)carve((size_t)512 * 512 * 2);
  h16* f1P   = (h16*)carve((size_t)256 * 1024 * 2);
  h16* f2P   = (h16*)carve((size_t)1024 * 256 * 2);
  h16* FR16  = (h16*)carve((size_t)131072 * 256 * 2);
  h16* FT16  = (h16*)carve((size_t)131072 * 256 * 2);
  float* rowStat = (float*)carve((size_t)131072 * 2 * 4);
  float* rRw = (float*)carve((size_t)131072 * 4);
  float* rTw = (float*)carve((size_t)131072 * 4);
  h16* Y16    = (h16*)carve((size_t)8192 * 512 * 2);
  h16* Zphi16 = (h16*)carve((size_t)8192 * 512 * 2);
  h16* Z16    = (h16*)carve((size_t)8192 * 256 * 2);
  h16* QZ16   = (h16*)carve((size_t)8192 * 256 * 2);
  h16* Zp16   = (h16*)carve((size_t)8192 * 256 * 2);
  h16* H16    = (h16*)carve((size_t)8192 * 1024 * 2);
  h16* Zp2    = (h16*)carve((size_t)8192 * 256 * 2);
  h16* KZ16   = (h16*)carve((size_t)8192 * 256 * 2);
  h16* VZ16   = (h16*)carve((size_t)8192 * 256 * 2);
  float* gammaB = (float*)carve(8 * 4);

  k_pack<<<(256 * 256 + 255) / 256, 256, 0, stream>>>(q_w, qP, 256, 256);
  k_pack<<<(256 * 256 + 255) / 256, 256, 0, stream>>>(k_w, kP, 256, 256);
  k_pack<<<(256 * 256 + 255) / 256, 256, 0, stream>>>(v_w, vP, 256, 256);
  k_pack<<<(512 * 512 + 255) / 256, 256, 0, stream>>>(phi_w, phiP, 512, 512);
  k_pack<<<(256 * 1024 + 255) / 256, 256, 0, stream>>>(fc1_w, f1P, 256, 1024);
  k_pack<<<(1024 * 256 + 255) / 256, 256, 0, stream>>>(fc2_w, f2P, 1024, 256);
  k_gamma<<<8, 256, 0, stream>>>(r_R, r_T, g1_w, g1_b, g2_w, g2_b, gammaB);
  k_winprep<<<2048, 256, 0, stream>>>(F_R, F_T, r_R, r_T, FR16, FT16, rowStat, rRw, rTw);
  k_psiA<<<2048, 256, 0, stream>>>(FR16, FT16, rowStat, psi_w, psi_b, np_g, np_b, Y16);
  k_gemm<0><<<dim3(4, 128), 256, 0, stream>>>(Y16, phiP, phi_b, Zphi16, 8192, 512, 512);
  k_Zln<<<2048, 128, 0, stream>>>(Zphi16, nz_g, nz_b, Z16);
  k_gemm<0><<<dim3(2, 128), 256, 0, stream>>>(Z16, qP, q_b, QZ16, 8192, 256, 256);
  k_xattn<<<2048, 256, 0, stream>>>(FR16, FT16, kP, vP, k_b, v_b, QZ16, Z16,
                                    rRw, rTw, nzf_g, nzf_b, Zp16);
  k_gemm<1><<<dim3(8, 128), 256, 0, stream>>>(Zp16, f1P, fc1_b, H16, 8192, 256, 1024);
  k_gemm<0><<<dim3(2, 128), 256, 0, stream>>>(H16, f2P, fc2_b, Zp2, 8192, 1024, 256);
  k_gemm<0><<<dim3(2, 128), 256, 0, stream>>>(Zp2, kP, k_b, KZ16, 8192, 256, 256);
  k_gemm<0><<<dim3(2, 128), 256, 0, stream>>>(Zp2, vP, v_b, VZ16, 8192, 256, 256);
  k_sattn<<<2048, 256, 0, stream>>>(FR16, qP, q_b, KZ16, VZ16, rTw, nr_g, nr_b, gammaB, out, 0);
  k_sattn<<<2048, 256, 0, stream>>>(FT16, qP, q_b, KZ16, VZ16, rRw, nt_g, nt_b, gammaB, out, 1);
}
